// MoE_CoKin_TransformerBlock_59974923321728
// MI455X (gfx1250) — matrix operand order repaired and checked by reading, not compiled, not run on hardware
//
#include <hip/hip_runtime.h>

// ---------------------------------------------------------------------------
// MoE_CoKin transformer block for MI455X (gfx1250), wave32 + WMMA bf16.
// All heavy contractions -> v_wmma_f32_16x16x32_bf16 (fp32 accumulate).
// GEMM: 128x64 block tile, 8 waves, double-buffered LDS; BOTH panels staged
// with global_load_async_to_lds_b128 (ASYNCcnt). Weights pre-transposed to
// [N][K] bf16 once per launch so no per-tile transpose is ever needed.
// ---------------------------------------------------------------------------

typedef __bf16 bf16_t;
typedef __attribute__((ext_vector_type(8)))  __bf16 bf16x8;
typedef __attribute__((ext_vector_type(16))) __bf16 v16bf;
typedef __attribute__((ext_vector_type(8)))  float  v8f;

#define D_MODEL 512
#define NHEAD 8
#define HDIM 64
#define LQ 1024
#define NSC 512
#define BATCH 8
#define DN_ 256
#define NEXP 8
#define TOPK 2
#define HE_ 2048

__device__ inline v8f vzero() {
  v8f v;
#pragma unroll
  for (int i = 0; i < 8; ++i) v[i] = 0.f;
  return v;
}

__device__ inline v16bf make_frag(bf16x8 lo, bf16x8 hi) {
  v16bf r;
#pragma unroll
  for (int i = 0; i < 8; ++i) { r[i] = lo[i]; r[i + 8] = hi[i]; }
  return r;
}

__device__ inline v8f wmma_bf16(v16bf a, v16bf b, v8f c) {
  return __builtin_amdgcn_wmma_f32_16x16x32_bf16(false, a, false, b, (short)0, c,
                                                 false, false);
}

// A-fragment (16x32, M x K): lane m = lane&15; element e -> K = e + 16*hi
__device__ inline v16bf load_afrag(const bf16_t* base, int stride, int lane) {
  int m = lane & 15, hi = (lane >> 4) & 1;
  const bf16_t* p = base + (size_t)m * stride + hi * 16;
  return make_frag(*(const bf16x8*)p, *(const bf16x8*)(p + 8));
}

// B-fragment (32x16, K x N) from K-contiguous (transposed) storage:
// lane n = lane&15; element e -> K = e + 16*hi
__device__ inline v16bf load_bfrag(const bf16_t* bt, int stride, int lane) {
  int n = lane & 15, hi = (lane >> 4) & 1;
  const bf16_t* p = bt + (size_t)n * stride + hi * 16;
  return make_frag(*(const bf16x8*)p, *(const bf16x8*)(p + 8));
}

// flat LDS pointer -> byte offset within the workgroup LDS segment
__device__ inline unsigned lds_addr32(const void* p) {
  return (unsigned)(unsigned long long)p;
}

// CDNA5 async global -> LDS copy (16B per lane), tracked by ASYNCcnt.
__device__ inline void async_load_b128(unsigned ldsOff, const void* gaddr) {
  asm volatile("global_load_async_to_lds_b128 %0, %1, off"
               :
               : "v"(ldsOff), "v"(gaddr)
               : "memory");
}
__device__ inline void wait_asynccnt0() {
  asm volatile("s_wait_asynccnt 0x0" ::: "memory");
}

// ---------------------------------------------------------------------------
// fp32 -> bf16 conversion
// ---------------------------------------------------------------------------
__global__ __launch_bounds__(256) void cvt_kernel(const float* __restrict__ in,
                                                  bf16_t* __restrict__ out, long n) {
  long i = (long)blockIdx.x * 1024 + threadIdx.x;
#pragma unroll
  for (int k = 0; k < 4; ++k) {
    long j = i + (long)k * 256;
    if (j < n) out[j] = (bf16_t)in[j];
  }
}

// fp32 [z][K][N] -> bf16 [z][N][K]  (weight transpose, done once per launch)
__global__ __launch_bounds__(256) void cvt_t_kernel(const float* __restrict__ in,
                                                    bf16_t* __restrict__ out,
                                                    int K, int N) {
  long idx = (long)blockIdx.x * 256 + threadIdx.x;
  long tot = (long)K * N;
  if (idx < tot) {
    int k = (int)(idx / N), n = (int)(idx % N);
    out[(size_t)blockIdx.z * tot + (size_t)n * K + k] =
        (bf16_t)in[(size_t)blockIdx.z * tot + idx];
  }
}

// ---------------------------------------------------------------------------
// Noise embedding + AdaLN modulations + router top-2  (B=8, tiny)
// ---------------------------------------------------------------------------
__global__ __launch_bounds__(256) void noise_mods_kernel(
    const float* __restrict__ t, const float* __restrict__ ne_w1,
    const float* __restrict__ ne_b1, const float* __restrict__ ne_w2,
    const float* __restrict__ ne_b2, const float* __restrict__ ncsa_w,
    const float* __restrict__ ncsa_b, const float* __restrict__ moe_w,
    const float* __restrict__ moe_b, const float* __restrict__ router_w,
    float* __restrict__ nc_out, float* __restrict__ ncsa_shift,
    float* __restrict__ ncsa_scale, float* __restrict__ moe_shift,
    float* __restrict__ moe_scale, int* __restrict__ ti, float* __restrict__ tw) {
  __shared__ float temb[D_MODEL];
  __shared__ float h1[DN_];
  __shared__ float ncs[DN_];
  int b = blockIdx.x, tid = threadIdx.x;
  float tv = t[b];
  {
    float fr = __expf(-logf(10000.f) * (float)tid / 255.f);
    float ang = tv * fr;
    temb[tid] = __cosf(ang);
    temb[tid + 256] = __sinf(ang);
  }
  __syncthreads();
  {
    float s = ne_b1[tid];
    for (int d = 0; d < D_MODEL; ++d) s += temb[d] * ne_w1[d * DN_ + tid];
    h1[tid] = s / (1.f + __expf(-s));
  }
  __syncthreads();
  {
    float s = ne_b2[tid];
    for (int k = 0; k < DN_; ++k) s += h1[k] * ne_w2[k * DN_ + tid];
    ncs[tid] = s;
    nc_out[b * DN_ + tid] = s;
  }
  __syncthreads();
  for (int jj = tid; jj < 2 * D_MODEL; jj += 256) {
    float m1 = ncsa_b[jj], m2 = moe_b[jj];
    for (int k = 0; k < DN_; ++k) {
      float nk = ncs[k];
      m1 += nk * ncsa_w[k * 2 * D_MODEL + jj];
      m2 += nk * moe_w[k * 2 * D_MODEL + jj];
    }
    if (jj < D_MODEL) {
      ncsa_shift[b * D_MODEL + jj] = m1;
      moe_shift[b * D_MODEL + jj] = m2;
    } else {
      ncsa_scale[b * D_MODEL + jj - D_MODEL] = m1;
      moe_scale[b * D_MODEL + jj - D_MODEL] = m2;
    }
  }
  if (tid == 0) {
    float lg[NEXP];
    float mx = -1e30f;
    for (int e = 0; e < NEXP; ++e) {
      float s = 0.f;
      for (int k = 0; k < DN_; ++k) s += ncs[k] * router_w[k * NEXP + e];
      lg[e] = s;
      mx = fmaxf(mx, s);
    }
    float se = 0.f;
    for (int e = 0; e < NEXP; ++e) { lg[e] = __expf(lg[e] - mx); se += lg[e]; }
    for (int e = 0; e < NEXP; ++e) lg[e] /= se;
    int i0 = 0;
    for (int e = 1; e < NEXP; ++e) if (lg[e] > lg[i0]) i0 = e;
    int i1 = (i0 == 0) ? 1 : 0;
    for (int e = 0; e < NEXP; ++e)
      if (e != i0 && lg[e] > lg[i1]) i1 = e;
    float ss = fmaxf(lg[i0] + lg[i1], 1e-8f);
    ti[b * TOPK] = i0; ti[b * TOPK + 1] = i1;
    tw[b * TOPK] = lg[i0] / ss; tw[b * TOPK + 1] = lg[i1] / ss;
  }
}

// ---------------------------------------------------------------------------
// LayerNorm (affine OR AdaLN) -> bf16
// ---------------------------------------------------------------------------
__global__ __launch_bounds__(256) void ln_kernel(
    const float* __restrict__ X, bf16_t* __restrict__ Out,
    const float* __restrict__ g, const float* __restrict__ bta,
    const float* __restrict__ shift, const float* __restrict__ scale, int L) {
  __shared__ float red[256];
  __shared__ float mu_s, rs_s;
  int row = blockIdx.x, tid = threadIdx.x;
  int bi = row / L;
  const float* xr = X + (size_t)row * D_MODEL;
  float v0 = xr[tid], v1 = xr[tid + 256];
  red[tid] = v0 + v1;
  __syncthreads();
  for (int s = 128; s > 0; s >>= 1) {
    if (tid < s) red[tid] += red[tid + s];
    __syncthreads();
  }
  if (tid == 0) mu_s = red[0] * (1.f / (float)D_MODEL);
  __syncthreads();
  float mu = mu_s;
  float d0 = v0 - mu, d1 = v1 - mu;
  red[tid] = d0 * d0 + d1 * d1;
  __syncthreads();
  for (int s = 128; s > 0; s >>= 1) {
    if (tid < s) red[tid] += red[tid + s];
    __syncthreads();
  }
  if (tid == 0) rs_s = rsqrtf(red[0] * (1.f / (float)D_MODEL) + 1e-5f);
  __syncthreads();
  float rs = rs_s;
#pragma unroll
  for (int p = 0; p < 2; ++p) {
    int j = tid + p * 256;
    float y = ((p ? v1 : v0) - mu) * rs;
    if (g) y = y * g[j] + bta[j];
    if (shift) y = y * (1.f + scale[bi * D_MODEL + j]) + shift[bi * D_MODEL + j];
    Out[(size_t)row * D_MODEL + j] = (bf16_t)y;
  }
}

// ---------------------------------------------------------------------------
// Generic batched WMMA GEMM: C[z] = A[z/zDiv] @ B[sel(z)]^T-stored (+bias)(+res)
// B is stored K-contiguous: BT[n*ldb + k]  (ldb == Kd stride)
// modes: 0 = bf16 out (+bias)
//        1 = f32 out  (+bias + residual)
//        2 = SwiGLU: out = (acc_v+bias_v) * silu(acc_g+bias_g), bf16 out
//            (gate rows at n + gateOff in BT, gate bias at gn + gateOff)
//        3 = f32 out  (+bias)
// 128x64 block tile, K-step 32, 8 waves (4x2), double-buffered LDS, all
// panel staging via global_load_async_to_lds_b128.
// ---------------------------------------------------------------------------
#define BM 128
#define BN 64
#define BKS 32
#define LDS_S 40

__global__ __launch_bounds__(256) void gemm_wmma_kernel(
    const bf16_t* __restrict__ A, const bf16_t* __restrict__ Bm,
    const float* __restrict__ bias, void* __restrict__ Cout,
    const float* __restrict__ Res, int M, int N, int Kd, int lda, int ldb,
    int ldc, long strideAz, int zDivA, long strideBz, long strideCz,
    long strideResz, const int* __restrict__ bsel, long selStrideB,
    long selStrideBias, int mode, int gateOff) {
  __shared__ bf16_t As[2][BM * LDS_S];
  __shared__ bf16_t Bt[2][BN * LDS_S];
  __shared__ bf16_t Btg[2][BN * LDS_S];

  int z = blockIdx.z;
  const bf16_t* Az = A + (size_t)(z / zDivA) * strideAz;
  int sel = bsel ? bsel[z] : z;
  const bf16_t* Bz =
      Bm + (bsel ? (size_t)sel * selStrideB : (size_t)z * strideBz);
  const float* biasz =
      bias ? bias + (bsel ? (size_t)sel * selStrideBias : 0) : nullptr;

  int m0 = blockIdx.y * BM, n0 = blockIdx.x * BN;
  int tid = threadIdx.x, lane = tid & 31, wv = tid >> 5;
  int wm = wv >> 1, wn = wv & 1;  // 4 x 2 wave grid, 32x32 per wave
  bool swi = (mode == 2);

  v8f acc[2][2], accg[2][2];
#pragma unroll
  for (int i = 0; i < 2; ++i)
#pragma unroll
    for (int j = 0; j < 2; ++j) { acc[i][j] = vzero(); accg[i][j] = vzero(); }

  // stage both panels for K-step k0 into buffer buf (pure async -> LDS)
  auto stage = [&](int buf, int k0) {
    // A panel 128x32: 512 16B-chunks, 2 per thread
#pragma unroll
    for (int it = 0; it < 2; ++it) {
      int c = it * 256 + tid;
      int r = c >> 2, cc = (c & 3) * 8;
      async_load_b128(lds_addr32(&As[buf][r * LDS_S + cc]),
                      &Az[(size_t)(m0 + r) * lda + k0 + cc]);
    }
    // B panel: 64 N-rows x 32 K: 256 16B-chunks, 1 per thread
    {
      int j = tid >> 2, kc = (tid & 3) * 8;
      async_load_b128(lds_addr32(&Bt[buf][j * LDS_S + kc]),
                      &Bz[(size_t)(n0 + j) * ldb + k0 + kc]);
      if (swi)
        async_load_b128(lds_addr32(&Btg[buf][j * LDS_S + kc]),
                        &Bz[(size_t)(n0 + j + gateOff) * ldb + k0 + kc]);
    }
  };

  auto compute = [&](int buf) {
    v16bf a0 = load_afrag(&As[buf][(wm * 32) * LDS_S], LDS_S, lane);
    v16bf a1 = load_afrag(&As[buf][(wm * 32 + 16) * LDS_S], LDS_S, lane);
    v16bf b0 = load_bfrag(&Bt[buf][(wn * 32) * LDS_S], LDS_S, lane);
    v16bf b1 = load_bfrag(&Bt[buf][(wn * 32 + 16) * LDS_S], LDS_S, lane);
    acc[0][0] = wmma_bf16(a0, b0, acc[0][0]);
    acc[0][1] = wmma_bf16(a0, b1, acc[0][1]);
    acc[1][0] = wmma_bf16(a1, b0, acc[1][0]);
    acc[1][1] = wmma_bf16(a1, b1, acc[1][1]);
    if (swi) {
      v16bf g0 = load_bfrag(&Btg[buf][(wn * 32) * LDS_S], LDS_S, lane);
      v16bf g1 = load_bfrag(&Btg[buf][(wn * 32 + 16) * LDS_S], LDS_S, lane);
      accg[0][0] = wmma_bf16(a0, g0, accg[0][0]);
      accg[0][1] = wmma_bf16(a0, g1, accg[0][1]);
      accg[1][0] = wmma_bf16(a1, g0, accg[1][0]);
      accg[1][1] = wmma_bf16(a1, g1, accg[1][1]);
    }
  };

  stage(0, 0);
  wait_asynccnt0();
  __syncthreads();
  int buf = 0;
  for (int k0 = 0; k0 < Kd; k0 += BKS) {
    int nk = k0 + BKS;
    if (nk < Kd) stage(buf ^ 1, nk);
    // prefetch panels two K-steps ahead toward L2/L0
    int pk = k0 + 2 * BKS;
    if (pk < Kd) {
      int r = tid >> 2, cc = (tid & 3) * 8;
      __builtin_prefetch(&Az[(size_t)(m0 + r) * lda + pk + cc], 0, 1);
      int j = tid >> 2, kc = (tid & 3) * 8;
      __builtin_prefetch(&Bz[(size_t)(n0 + j) * ldb + pk + kc], 0, 1);
    }
    compute(buf);
    wait_asynccnt0();
    __syncthreads();
    buf ^= 1;
  }

  // epilogue
  int nl = lane & 15, hi = lane >> 4;
#pragma unroll
  for (int i = 0; i < 2; ++i)
#pragma unroll
    for (int j = 0; j < 2; ++j) {
#pragma unroll
      for (int e = 0; e < 8; ++e) {
        int gm = m0 + wm * 32 + i * 16 + e + 8 * hi;
        int gn = n0 + wn * 32 + j * 16 + nl;
        float v = acc[i][j][e];
        if (biasz) v += biasz[gn];
        size_t cidx = (size_t)gm * ldc + gn;
        if (mode == 0) {
          ((bf16_t*)Cout)[(size_t)z * strideCz + cidx] = (bf16_t)v;
        } else if (mode == 1) {
          float* Co = (float*)Cout + (size_t)z * strideCz;
          const float* Rz = Res + (size_t)z * strideResz;
          Co[cidx] = v + Rz[cidx];
        } else if (mode == 2) {
          float gg = accg[i][j][e] + (biasz ? biasz[gn + gateOff] : 0.f);
          float hsw = v * (gg / (1.f + __expf(-gg)));
          ((bf16_t*)Cout)[(size_t)z * strideCz + cidx] = (bf16_t)hsw;
        } else {
          ((float*)Cout)[(size_t)z * strideCz + cidx] = v;
        }
      }
    }
}

// ---------------------------------------------------------------------------
// naive bf16 transpose: [B, Lr, 512] -> [B, 512, Lr]
// ---------------------------------------------------------------------------
__global__ __launch_bounds__(256) void transpose_kernel(
    const bf16_t* __restrict__ in, bf16_t* __restrict__ out, int Lr) {
  long idx = (long)blockIdx.x * 256 + threadIdx.x;
  long tot = (long)Lr * D_MODEL;
  if (idx < tot) {
    int l = (int)(idx >> 9), d = (int)(idx & 511);
    out[((size_t)blockIdx.z * D_MODEL + d) * Lr + l] =
        in[((size_t)blockIdx.z * Lr + l) * D_MODEL + d];
  }
}

// ---------------------------------------------------------------------------
// Attention: one wave per (b, h, 16-query strip). Scores + PV via WMMA.
// Q,Km: [B, L, 512] bf16 (head h -> cols h*64..); Vt: [B, 512, Lk] bf16.
// dynamic LDS: 16*Lk floats (scores; rows repacked in-place to bf16).
// ---------------------------------------------------------------------------
__global__ __launch_bounds__(32) void attn_wmma_kernel(
    const bf16_t* __restrict__ Q, const bf16_t* __restrict__ Km,
    const bf16_t* __restrict__ Vt, bf16_t* __restrict__ O, int Lq, int Lk,
    float scale) {
  extern __shared__ float smem[];  // 16 * Lk floats
  int m0 = blockIdx.x * 16, h = blockIdx.y, b = blockIdx.z;
  int lane = threadIdx.x & 31;
  int nl = lane & 15, hi = lane >> 4;

  const bf16_t* qbase = Q + ((size_t)b * Lq + m0) * D_MODEL + h * HDIM;
  v16bf a0 = load_afrag(qbase, D_MODEL, lane);       // hd 0..31
  v16bf a1 = load_afrag(qbase + 32, D_MODEL, lane);  // hd 32..63

  const bf16_t* kbase = Km + (size_t)b * Lk * D_MODEL + h * HDIM;
  for (int kt = 0; kt < Lk / 16; ++kt) {
    const bf16_t* kp = kbase + (size_t)(kt * 16 + nl) * D_MODEL + hi * 16;
    v16bf b0 = make_frag(*(const bf16x8*)kp, *(const bf16x8*)(kp + 8));
    v16bf b1 = make_frag(*(const bf16x8*)(kp + 32), *(const bf16x8*)(kp + 40));
    v8f acc = vzero();
    acc = wmma_bf16(a0, b0, acc);
    acc = wmma_bf16(a1, b1, acc);
#pragma unroll
    for (int e = 0; e < 8; ++e)
      smem[(size_t)(e + 8 * hi) * Lk + kt * 16 + nl] = acc[e] * scale;
  }
  __syncthreads();

  if (lane < 16) {
    float* row = smem + (size_t)lane * Lk;
    float mx = -1e30f;
    for (int c = 0; c < Lk; ++c) mx = fmaxf(mx, row[c]);
    float sum = 0.f;
    for (int c = 0; c < Lk; ++c) {
      float e = __expf(row[c] - mx);
      row[c] = e;
      sum += e;
    }
    float inv = 1.f / sum;
    unsigned* pk = (unsigned*)row;
    for (int c = 0; c < Lk; c += 2) {
      union { __bf16 hh[2]; unsigned u; } cv;
      cv.hh[0] = (bf16_t)(row[c] * inv);
      cv.hh[1] = (bf16_t)(row[c + 1] * inv);
      pk[c >> 1] = cv.u;
    }
  }
  __syncthreads();

  v8f oacc[4];
#pragma unroll
  for (int nt = 0; nt < 4; ++nt) oacc[nt] = vzero();
  const bf16_t* prow = (const bf16_t*)(smem + (size_t)nl * Lk);
  const bf16_t* vbase = Vt + ((size_t)b * D_MODEL + h * HDIM) * Lk;
  for (int kt = 0; kt < Lk / 32; ++kt) {
    const bf16_t* pp = prow + kt * 32 + hi * 16;
    v16bf ap = make_frag(*(const bf16x8*)pp, *(const bf16x8*)(pp + 8));
#pragma unroll
    for (int nt = 0; nt < 4; ++nt) {
      const bf16_t* vp = vbase + (size_t)(nt * 16 + nl) * Lk + kt * 32 + hi * 16;
      v16bf bv = make_frag(*(const bf16x8*)vp, *(const bf16x8*)(vp + 8));
      oacc[nt] = wmma_bf16(ap, bv, oacc[nt]);
    }
  }
  bf16_t* ob = O + ((size_t)b * Lq + m0) * D_MODEL + h * HDIM;
#pragma unroll
  for (int nt = 0; nt < 4; ++nt)
#pragma unroll
    for (int e = 0; e < 8; ++e)
      ob[(size_t)(e + 8 * hi) * D_MODEL + nt * 16 + nl] = (bf16_t)oacc[nt][e];
}

// ---------------------------------------------------------------------------
// final combine: out = x2 + tw[2b]*eo[2b] + tw[2b+1]*eo[2b+1]
// ---------------------------------------------------------------------------
__global__ __launch_bounds__(256) void combine_kernel(
    const float* __restrict__ x2, const float* __restrict__ eo,
    const float* __restrict__ tw, float* __restrict__ out) {
  long idx = (long)blockIdx.x * 256 + threadIdx.x;
  int b = (int)(idx >> 19);          // L*D = 2^19
  long off = idx & ((1L << 19) - 1);
  out[idx] = x2[idx] + tw[2 * b] * eo[((long)(2 * b) << 19) + off] +
             tw[2 * b + 1] * eo[((long)(2 * b + 1) << 19) + off];
}

// ---------------------------------------------------------------------------
// host side
// ---------------------------------------------------------------------------
extern "C" void kernel_launch(void* const* d_in, const int* in_sizes, int n_in,
                              void* d_out, int out_size, void* d_ws,
                              size_t ws_size, hipStream_t stream) {
  (void)in_sizes; (void)n_in; (void)out_size; (void)ws_size;
  const float* x = (const float*)d_in[0];
  const float* scene = (const float*)d_in[1];
  const float* t = (const float*)d_in[2];
  const float* sn_g = (const float*)d_in[3];
  const float* sn_b = (const float*)d_in[4];
  const float* ca_wq = (const float*)d_in[5];  const float* ca_bq = (const float*)d_in[6];
  const float* ca_wk = (const float*)d_in[7];  const float* ca_bk = (const float*)d_in[8];
  const float* ca_wv = (const float*)d_in[9];  const float* ca_bv = (const float*)d_in[10];
  const float* ca_wo = (const float*)d_in[11]; const float* ca_bo = (const float*)d_in[12];
  const float* sa_wq = (const float*)d_in[13]; const float* sa_bq = (const float*)d_in[14];
  const float* sa_wk = (const float*)d_in[15]; const float* sa_bk = (const float*)d_in[16];
  const float* sa_wv = (const float*)d_in[17]; const float* sa_bv = (const float*)d_in[18];
  const float* sa_wo = (const float*)d_in[19]; const float* sa_bo = (const float*)d_in[20];
  const float* ne_w1 = (const float*)d_in[21]; const float* ne_b1 = (const float*)d_in[22];
  const float* ne_w2 = (const float*)d_in[23]; const float* ne_b2 = (const float*)d_in[24];
  const float* ncsa_w = (const float*)d_in[25]; const float* ncsa_b = (const float*)d_in[26];
  const float* moe_w = (const float*)d_in[27];  const float* moe_b = (const float*)d_in[28];
  const float* router_w = (const float*)d_in[29];
  const float* fc_in_w = (const float*)d_in[30]; const float* fc_in_b = (const float*)d_in[31];
  const float* fc_out_w = (const float*)d_in[32]; const float* fc_out_b = (const float*)d_in[33];

  char* ws = (char*)d_ws;
  size_t off = 0;
  auto alloc = [&](size_t bytes) -> void* {
    off = (off + 255) & ~(size_t)255;
    void* p = ws + off;
    off += bytes;
    return p;
  };
  const size_t BLD = (size_t)BATCH * LQ * D_MODEL;    // 4,194,304
  const size_t BSD = (size_t)BATCH * NSC * D_MODEL;   // 2,097,152
  const size_t WSQ = (size_t)D_MODEL * D_MODEL;       // 262,144

  bf16_t* w8[8];                                      // [N][K] transposed bf16
  for (int i = 0; i < 8; ++i) w8[i] = (bf16_t*)alloc(WSQ * 2);
  bf16_t* w_fcin = (bf16_t*)alloc((size_t)NEXP * D_MODEL * 2 * HE_ * 2);  // [E][4096][512]
  bf16_t* w_fcout = (bf16_t*)alloc((size_t)NEXP * HE_ * D_MODEL * 2);    // [E][512][2048]
  bf16_t* actb = (bf16_t*)alloc(BLD * 2);
  bf16_t* sceneb = (bf16_t*)alloc(BSD * 2);
  bf16_t* qb = (bf16_t*)alloc(BLD * 2);
  bf16_t* kb = (bf16_t*)alloc(BLD * 2);
  bf16_t* vb = (bf16_t*)alloc(BLD * 2);
  bf16_t* vt = (bf16_t*)alloc(BLD * 2);
  bf16_t* ao = (bf16_t*)alloc(BLD * 2);
  float* x1 = (float*)alloc(BLD * 4);
  float* x2 = (float*)alloc(BLD * 4);
  bf16_t* hid = (bf16_t*)alloc((size_t)BATCH * TOPK * LQ * HE_ * 2);
  float* eo = (float*)alloc((size_t)BATCH * TOPK * LQ * D_MODEL * 4);
  float* ncbuf = (float*)alloc(BATCH * DN_ * 4);
  float* ncsa_sh = (float*)alloc(BATCH * D_MODEL * 4);
  float* ncsa_sc = (float*)alloc(BATCH * D_MODEL * 4);
  float* moe_sh = (float*)alloc(BATCH * D_MODEL * 4);
  float* moe_sc = (float*)alloc(BATCH * D_MODEL * 4);
  int* ti = (int*)alloc(BATCH * TOPK * 4);
  float* tw = (float*)alloc(BATCH * TOPK * 4);

  // weight transposes (fp32 [K][N] -> bf16 [N][K]); activations: plain cvt
  auto cvtT = [&](const float* src, bf16_t* dst, int K, int N, int Z) {
    long tot = (long)K * N;
    cvt_t_kernel<<<dim3((unsigned)((tot + 255) / 256), 1, Z), 256, 0, stream>>>(
        src, dst, K, N);
  };
  cvtT(ca_wq, w8[0], D_MODEL, D_MODEL, 1); cvtT(ca_wk, w8[1], D_MODEL, D_MODEL, 1);
  cvtT(ca_wv, w8[2], D_MODEL, D_MODEL, 1); cvtT(ca_wo, w8[3], D_MODEL, D_MODEL, 1);
  cvtT(sa_wq, w8[4], D_MODEL, D_MODEL, 1); cvtT(sa_wk, w8[5], D_MODEL, D_MODEL, 1);
  cvtT(sa_wv, w8[6], D_MODEL, D_MODEL, 1); cvtT(sa_wo, w8[7], D_MODEL, D_MODEL, 1);
  cvtT(fc_in_w, w_fcin, D_MODEL, 2 * HE_, NEXP);
  cvtT(fc_out_w, w_fcout, HE_, D_MODEL, NEXP);
  cvt_kernel<<<dim3((unsigned)((BSD + 1023) / 1024)), 256, 0, stream>>>(
      scene, sceneb, (long)BSD);

  noise_mods_kernel<<<BATCH, 256, 0, stream>>>(
      t, ne_w1, ne_b1, ne_w2, ne_b2, ncsa_w, ncsa_b, moe_w, moe_b, router_w,
      ncbuf, ncsa_sh, ncsa_sc, moe_sh, moe_sc, ti, tw);

  // --- scene cross-attention ---
  ln_kernel<<<BATCH * LQ, 256, 0, stream>>>(x, actb, sn_g, sn_b, nullptr, nullptr, LQ);
  gemm_wmma_kernel<<<dim3(D_MODEL / BN, (BATCH * LQ) / BM, 1), 256, 0, stream>>>(
      actb, w8[0], ca_bq, qb, nullptr, BATCH * LQ, D_MODEL, D_MODEL, D_MODEL,
      D_MODEL, D_MODEL, 0, 1, 0, 0, 0, nullptr, 0, 0, 0, 0);
  gemm_wmma_kernel<<<dim3(D_MODEL / BN, (BATCH * NSC) / BM, 1), 256, 0, stream>>>(
      sceneb, w8[1], ca_bk, kb, nullptr, BATCH * NSC, D_MODEL, D_MODEL, D_MODEL,
      D_MODEL, D_MODEL, 0, 1, 0, 0, 0, nullptr, 0, 0, 0, 0);
  gemm_wmma_kernel<<<dim3(D_MODEL / BN, (BATCH * NSC) / BM, 1), 256, 0, stream>>>(
      sceneb, w8[2], ca_bv, vb, nullptr, BATCH * NSC, D_MODEL, D_MODEL, D_MODEL,
      D_MODEL, D_MODEL, 0, 1, 0, 0, 0, nullptr, 0, 0, 0, 0);
  transpose_kernel<<<dim3((unsigned)(((long)NSC * D_MODEL + 255) / 256), 1, BATCH),
                     256, 0, stream>>>(vb, vt, NSC);
  attn_wmma_kernel<<<dim3(LQ / 16, NHEAD, BATCH), 32, 16 * NSC * 4, stream>>>(
      qb, kb, vt, ao, LQ, NSC, 0.125f);
  gemm_wmma_kernel<<<dim3(D_MODEL / BN, (BATCH * LQ) / BM, 1), 256, 0, stream>>>(
      ao, w8[3], ca_bo, x1, x, BATCH * LQ, D_MODEL, D_MODEL, D_MODEL, D_MODEL,
      D_MODEL, 0, 1, 0, 0, 0, nullptr, 0, 0, 1, 0);

  // --- noise-conditioned self-attention ---
  ln_kernel<<<BATCH * LQ, 256, 0, stream>>>(x1, actb, nullptr, nullptr, ncsa_sh,
                                            ncsa_sc, LQ);
  gemm_wmma_kernel<<<dim3(D_MODEL / BN, (BATCH * LQ) / BM, 1), 256, 0, stream>>>(
      actb, w8[4], sa_bq, qb, nullptr, BATCH * LQ, D_MODEL, D_MODEL, D_MODEL,
      D_MODEL, D_MODEL, 0, 1, 0, 0, 0, nullptr, 0, 0, 0, 0);
  gemm_wmma_kernel<<<dim3(D_MODEL / BN, (BATCH * LQ) / BM, 1), 256, 0, stream>>>(
      actb, w8[5], sa_bk, kb, nullptr, BATCH * LQ, D_MODEL, D_MODEL, D_MODEL,
      D_MODEL, D_MODEL, 0, 1, 0, 0, 0, nullptr, 0, 0, 0, 0);
  gemm_wmma_kernel<<<dim3(D_MODEL / BN, (BATCH * LQ) / BM, 1), 256, 0, stream>>>(
      actb, w8[6], sa_bv, vb, nullptr, BATCH * LQ, D_MODEL, D_MODEL, D_MODEL,
      D_MODEL, D_MODEL, 0, 1, 0, 0, 0, nullptr, 0, 0, 0, 0);
  transpose_kernel<<<dim3((unsigned)(((long)LQ * D_MODEL + 255) / 256), 1, BATCH),
                     256, 0, stream>>>(vb, vt, LQ);
  attn_wmma_kernel<<<dim3(LQ / 16, NHEAD, BATCH), 32, 16 * LQ * 4, stream>>>(
      qb, kb, vt, ao, LQ, LQ, 0.125f);
  gemm_wmma_kernel<<<dim3(D_MODEL / BN, (BATCH * LQ) / BM, 1), 256, 0, stream>>>(
      ao, w8[7], sa_bo, x2, x1, BATCH * LQ, D_MODEL, D_MODEL, D_MODEL, D_MODEL,
      D_MODEL, 0, 1, 0, 0, 0, nullptr, 0, 0, 1, 0);

  // --- MoE pose branch ---
  ln_kernel<<<BATCH * LQ, 256, 0, stream>>>(x2, actb, nullptr, nullptr, moe_sh,
                                            moe_sc, LQ);
  // stage 1: hid[z] = swiglu(x_moe[b] @ fc_in_w[ti[z]] + b), z = b*2 + k
  // BT layout per expert: [4096 rows][512]; val rows n, gate rows n + HE_
  gemm_wmma_kernel<<<dim3(HE_ / BN, LQ / BM, BATCH * TOPK), 256, 0, stream>>>(
      actb, w_fcin, fc_in_b, hid, nullptr, LQ, HE_, D_MODEL, D_MODEL, D_MODEL,
      HE_, (long)LQ * D_MODEL, TOPK, 0, (long)LQ * HE_, 0, ti,
      (long)D_MODEL * 2 * HE_, 2 * HE_, 2, HE_);
  // stage 2: eo[z] = hid[z] @ fc_out_w[ti[z]] + b ; BT per expert [512][2048]
  gemm_wmma_kernel<<<dim3(D_MODEL / BN, LQ / BM, BATCH * TOPK), 256, 0, stream>>>(
      hid, w_fcout, fc_out_b, eo, nullptr, LQ, D_MODEL, HE_, HE_, HE_,
      D_MODEL, (long)LQ * HE_, 1, 0, (long)LQ * D_MODEL, 0, ti,
      (long)HE_ * D_MODEL, D_MODEL, 3, 0);
  combine_kernel<<<dim3((unsigned)(BLD / 256)), 256, 0, stream>>>(
      x2, eo, tw, (float*)d_out);
}
